// ExtraMsaStack_30391188586651
// MI455X (gfx1250) — compile-verified, matrix operand order checked by reading, not run on hardware
//
#include <hip/hip_runtime.h>

// ============================================================================
// ExtraMsaStack (1 block) on gfx1250: all large GEMMs via v_wmma_f32_16x16x32_bf16
// bf16 inputs / fp32 accumulate. Weights pre-packed on device into [N,K] bf16 so
// B fragments load exactly like A fragments (two 16B loads per lane, 16-bit
// A-matrix lane layout). GEMM v3: N-register-blocking (NT tiles/wave), A tile
// double-buffered through LDS via GLOBAL_LOAD_ASYNC_TO_LDS_B128 (ASYNCcnt),
// and global_prefetch_b8 on the B stream.
// ============================================================================

#define S_DIM 512
#define I_DIM 256
#define CE    64
#define CZ    128
#define HG    8
#define DG    8
#define HT    4
#define DT    32
#define COP   32
#define SI    (S_DIM * I_DIM)   // 131072
#define II    (I_DIM * I_DIM)   // 65536

typedef __attribute__((ext_vector_type(16))) __bf16 v16bf;
typedef __attribute__((ext_vector_type(8)))  __bf16 v8bf;
typedef __attribute__((ext_vector_type(8)))  float  v8f;
typedef unsigned short u16;

// ---------------- bf16 helpers (bit-level; storage is u16) ------------------
__device__ __forceinline__ u16 f2b(float f) {
  union { float f; unsigned u; } v; v.f = f;
  unsigned u = v.u;
  u += 0x7FFFu + ((u >> 16) & 1u);   // round-to-nearest-even
  return (u16)(u >> 16);
}
__device__ __forceinline__ float b2f(u16 h) {
  union { unsigned u; float f; } v; v.u = ((unsigned)h) << 16;
  return v.f;
}

// 16-bit A-fragment (16x32) lane layout: lane<16 holds row M=lane, K={0..7,16..23};
// lane>=16 holds row M=lane-16, K={8..15,24..31}. Two contiguous 16B loads.
__device__ __forceinline__ v16bf load_frag(const u16* __restrict__ base, long ld, int lane) {
  const int r  = lane & 15;
  const int kb = (lane >> 4) << 3;
  const u16* p = base + (long)r * ld + kb;
  v8bf lo = *(const v8bf*)p;
  v8bf hi = *(const v8bf*)(p + 16);
  v16bf f;
#pragma unroll
  for (int i = 0; i < 8; ++i) { f[i] = lo[i]; f[i + 8] = hi[i]; }
  return f;
}

// Same fragment gather but from an LDS-resident 16x32 tile (ld == 32).
__device__ __forceinline__ v16bf lds_frag(const u16* s, int lane) {
  const int r  = lane & 15;
  const int kb = (lane >> 4) << 3;
  const u16* p = s + r * 32 + kb;
  v8bf lo = *(const v8bf*)p;
  v8bf hi = *(const v8bf*)(p + 16);
  v16bf f;
#pragma unroll
  for (int i = 0; i < 8; ++i) { f[i] = lo[i]; f[i + 8] = hi[i]; }
  return f;
}

// Async global->LDS 16B copy (CDNA5; tracked by ASYNCcnt). LDS destination is
// the low 32 bits of the generic shared-pointer (flat->LDS truncates addr[31:0]).
__device__ __forceinline__ void async_stage(const u16* g, u16* l) {
  unsigned lds = (unsigned)(unsigned long long)l;
  asm volatile("global_load_async_to_lds_b128 %0, %1, off"
               :: "v"(lds), "v"(g) : "memory");
}
__device__ __forceinline__ void wait_async() {
  asm volatile("s_wait_asynccnt 0x0" ::: "memory");
}

// ============================================================================
// Generic batched WMMA GEMM:  C[b][M,N] = epi( alpha * A[b][M,K] * B[b][N,K]^T )
// 4 waves / block; each wave owns NT 16x16 output tiles along N, all waves share
// the block's 16-row M tile, double-buffered through LDS with async copies.
// epi modes: 0: t=a*acc+bias ; 1: t=(a*acc+bias)*rs[m] ; 2: t=a*acc*rs[m]+bias
// act: 0 none, 1 relu, 2 sigmoid.  Optional resid add (same layout as C).
// ============================================================================
template <int NT>
__global__ __launch_bounds__(128)
void wmma_gemm_kernel(const u16* __restrict__ A, const u16* __restrict__ B,
                      const float* __restrict__ bias, const float* __restrict__ rs,
                      const float* __restrict__ resid,
                      float* __restrict__ Cf, u16* __restrict__ Cb,
                      int M, int N, int K,
                      long lda, long ldb, long ldc,
                      long sA0, long sA1, long sB0, long sB1, long sC0, long sC1,
                      int nb1, float alpha, int act, int mode) {
  __shared__ u16 As[2][16 * 32];
  const int wave = threadIdx.x >> 5;
  const int lane = threadIdx.x & 31;
  const int tid  = threadIdx.x;
  const int tm = blockIdx.y;
  const int bz = blockIdx.z;
  const int b0 = bz / nb1, b1 = bz % nb1;
  const int tnb = (blockIdx.x * 4 + wave) * NT;
  const bool active = (tnb * 16) < N;

  const u16* Ab = A + (long)b0 * sA0 + (long)b1 * sA1 + (long)tm * 16 * lda;
  const u16* Bb[NT];
#pragma unroll
  for (int t = 0; t < NT; ++t) {
    const int tn = active ? (tnb + t) : 0;
    Bb[t] = B + (long)b0 * sB0 + (long)b1 * sB1 + (long)tn * 16 * ldb;
  }

  // Stage first 16x32 A slab into LDS slot 0 (one async 16B copy per lane<64).
  if (tid < 64) {
    const int r = tid >> 2, c = (tid & 3) << 3;
    async_stage(Ab + (long)r * lda + c, &As[0][r * 32 + c]);
  }
  wait_async();

  v8f acc[NT];
#pragma unroll
  for (int t = 0; t < NT; ++t)
#pragma unroll
    for (int e = 0; e < 8; ++e) acc[t][e] = 0.f;

  int slot = 0;
  for (int kk = 0; kk < K; kk += 32) {
    __syncthreads();
    if (kk + 32 < K && tid < 64) {           // async-prefetch next A slab
      const int r = tid >> 2, c = (tid & 3) << 3;
      async_stage(Ab + (long)r * lda + (kk + 32) + c, &As[slot ^ 1][r * 32 + c]);
    }
    v16bf a = lds_frag(&As[slot][0], lane);
#pragma unroll
    for (int t = 0; t < NT; ++t) {
      __builtin_prefetch(Bb[t] + kk + 64, 0, 3);   // -> global_prefetch_b8
      v16bf b = load_frag(Bb[t] + kk, ldb, lane);
      acc[t] = __builtin_amdgcn_wmma_f32_16x16x32_bf16(false, a, false, b, (short)0,
                                                       acc[t], false, false);
    }
    wait_async();                             // next slab resident before barrier
    slot ^= 1;
  }
  if (!active) return;

  // C/D layout: VGPR e: lanes 0-15 -> M=e, N=lane ; lanes 16-31 -> M=e+8, N=lane-16
  const int mbase = tm * 16 + ((lane >> 4) << 3);
  const long cbo = (long)b0 * sC0 + (long)b1 * sC1;
#pragma unroll
  for (int t = 0; t < NT; ++t) {
    const int n = (tnb + t) * 16 + (lane & 15);
    const float bn = bias ? bias[n] : 0.f;
#pragma unroll
    for (int e = 0; e < 8; ++e) {
      const int m = mbase + e;
      float v = alpha * acc[t][e];
      if (mode == 0)      v = v + bn;
      else if (mode == 1) v = (v + bn) * rs[m];
      else                v = v * rs[m] + bn;
      if (act == 1)      v = v > 0.f ? v : 0.f;
      else if (act == 2) v = 1.f / (1.f + __expf(-v));
      const long off = cbo + (long)m * ldc + n;
      if (resid) v += resid[off];
      if (Cb) Cb[off] = f2b(v);
      else    Cf[off] = v;
    }
  }
}

// ---------------- LayerNorm (rows of C=64 or 128), f32 -> bf16 --------------
__global__ __launch_bounds__(128)
void ln_rows_kernel(const float* __restrict__ x, const float* __restrict__ g,
                    const float* __restrict__ b, u16* __restrict__ out, int R, int C) {
  const int wave = threadIdx.x >> 5, lane = threadIdx.x & 31;
  const int row = blockIdx.x * 4 + wave;
  if (row >= R) return;
  const float* xr = x + (long)row * C;
  const int per = C >> 5;           // 2 (C=64) or 4 (C=128)
  float v[4];
  float s = 0.f;
  for (int i = 0; i < per; ++i) { v[i] = xr[lane + i * 32]; s += v[i]; }
  for (int o = 16; o; o >>= 1) s += __shfl_xor(s, o, 32);
  const float mu = s / (float)C;
  float q = 0.f;
  for (int i = 0; i < per; ++i) { float d = v[i] - mu; q += d * d; }
  for (int o = 16; o; o >>= 1) q += __shfl_xor(q, o, 32);
  const float inv = rsqrtf(q / (float)C + 1e-5f);
  for (int i = 0; i < per; ++i) {
    const int c = lane + i * 32;
    out[(long)row * C + c] = f2b((v[i] - mu) * inv * g[c] + b[c]);
  }
}

// ---------------- pack weight f32[K,N] -> bf16[Npad,K] (transpose) ----------
__global__ __launch_bounds__(256)
void pack_w_kernel(const float* __restrict__ W, int K, int N, int Npad,
                   u16* __restrict__ out) {
  const int idx = blockIdx.x * 256 + threadIdx.x;
  if (idx >= Npad * K) return;
  const int n = idx / K, k = idx % K;
  out[idx] = f2b(n < N ? W[(long)k * N + n] : 0.f);
}

// ---------------- mean over s: msa(s,i,c) -> (i,c) --------------------------
__global__ __launch_bounds__(256)
void mean_s_kernel(const float* __restrict__ msa, float* __restrict__ out) {
  const int idx = blockIdx.x * 256 + threadIdx.x;
  if (idx >= I_DIM * CE) return;
  const int i = idx / CE, c = idx % CE;
  float s = 0.f;
  for (int ss = 0; ss < S_DIM; ++ss) s += msa[((long)ss * I_DIM + i) * CE + c];
  out[idx] = s * (1.f / (float)S_DIM);
}

// ------------- global-attn scores: (i,h,s) = Q(i,h,:).K(s,i,h,:)/sqrt(8) ----
__global__ __launch_bounds__(256)
void ga_score_kernel(const float* __restrict__ Q, const u16* __restrict__ Kb,
                     const float* __restrict__ mask, float* __restrict__ sc) {
  const int idx = blockIdx.x * 256 + threadIdx.x;
  if (idx >= S_DIM * I_DIM * HG) return;
  const int h = idx & 7, i = (idx >> 3) & 255, s = idx >> 11;
  float a = 0.f;
#pragma unroll
  for (int d = 0; d < DG; ++d)
    a += Q[i * CE + h * DG + d] * b2f(Kb[((long)s * I_DIM + i) * CE + h * DG + d]);
  const float m = mask[s * I_DIM + i];
  sc[((long)i * HG + h) * S_DIM + s] = a * 0.35355339f + (1.f - m) * (-1e9f);
}

// ---------------- softmax over 512 (wave per row) ---------------------------
__global__ __launch_bounds__(128)
void softmax512_kernel(float* __restrict__ sc, int rows) {
  const int wave = threadIdx.x >> 5, lane = threadIdx.x & 31;
  const int row = blockIdx.x * 4 + wave;
  if (row >= rows) return;
  float* r = sc + (long)row * 512;
  float v[16]; float mx = -3.0e38f;
#pragma unroll
  for (int t = 0; t < 16; ++t) { v[t] = r[lane + t * 32]; mx = fmaxf(mx, v[t]); }
  for (int o = 16; o; o >>= 1) mx = fmaxf(mx, __shfl_xor(mx, o, 32));
  float s = 0.f;
#pragma unroll
  for (int t = 0; t < 16; ++t) { v[t] = __expf(v[t] - mx); s += v[t]; }
  for (int o = 16; o; o >>= 1) s += __shfl_xor(s, o, 32);
  const float inv = 1.f / s;
#pragma unroll
  for (int t = 0; t < 16; ++t) r[lane + t * 32] = v[t] * inv;
}

// ------------- global-attn weighted V: vals(i,h,d) = sum_s attn*V -----------
__global__ __launch_bounds__(256)
void ga_wv_kernel(const float* __restrict__ attn, const u16* __restrict__ V,
                  float* __restrict__ vals) {
  const int idx = blockIdx.x * 256 + threadIdx.x;
  if (idx >= I_DIM * HG * DG) return;
  const int d = idx & 7, h = (idx >> 3) & 7, i = idx >> 6;
  float a = 0.f;
  for (int s = 0; s < S_DIM; ++s)
    a += attn[((long)i * HG + h) * S_DIM + s] *
         b2f(V[((long)s * I_DIM + i) * CE + h * DG + d]);
  vals[i * CE + h * DG + d] = a;
}

// ------------- gate: gated(s,i,c) = sigmoid(G)(s,i,c) * vals(i,c) -----------
__global__ __launch_bounds__(256)
void ga_gate_kernel(const u16* __restrict__ G, const float* __restrict__ vals,
                    u16* __restrict__ out) {
  const int idx = blockIdx.x * 256 + threadIdx.x;
  if (idx >= SI * CE) return;
  const int c = idx & 63, r = idx >> 6, i = r & 255;
  out[idx] = f2b(b2f(G[idx]) * vals[i * CE + c]);
}

// ------------- OPM: (s*I+i, c) bf16 -> ((i*32+c), s) bf16 -------------------
__global__ __launch_bounds__(256)
void opm_tpack_kernel(const u16* __restrict__ A, u16* __restrict__ At) {
  const int idx = blockIdx.x * 256 + threadIdx.x;
  if (idx >= 8192 * S_DIM) return;
  const int s = idx & 511, r = idx >> 9;      // r = i*32+c
  const int i = r >> 5, c = r & 31;
  At[idx] = A[((long)s * I_DIM + i) * COP + c];
}

// ------------- OPM: ((i,c),(j,d)) bf16 -> ((i,j),(c,d)) bf16 ----------------
__global__ __launch_bounds__(256)
void opm_repack_kernel(const u16* __restrict__ O, u16* __restrict__ O2) {
  const int idx = blockIdx.x * 256 + threadIdx.x;
  if (idx >= II * COP * COP) return;
  const int f = idx & 1023, m = idx >> 10;    // m=(i*256+j), f=(c*32+d)
  const int i = m >> 8, j = m & 255, c = f >> 5, d = f & 31;
  O2[idx] = O[(long)(i * 32 + c) * 8192 + (j * 32 + d)];
}

// ------------- OPM norm: 1 / max(1, sum_s mask(s,i)*mask(s,j)) --------------
__global__ __launch_bounds__(256)
void norm_inv_kernel(const float* __restrict__ mask, float* __restrict__ ninv) {
  const int idx = blockIdx.x * 256 + threadIdx.x;
  if (idx >= II) return;
  const int i = idx >> 8, j = idx & 255;
  float s = 0.f;
  for (int ss = 0; ss < S_DIM; ++ss)
    s += mask[ss * I_DIM + i] * mask[ss * I_DIM + j];
  ninv[idx] = 1.f / fmaxf(s, 1.f);
}

// ------------- triangle softmax over k=256 (wave per row), bf16 in/out ------
// rows ordered (bi, h, m): start: bi=i,m=j; end: bi=j,m=i
__global__ __launch_bounds__(128)
void tri_softmax_kernel(u16* __restrict__ S, const float* __restrict__ Bh,
                        const float* __restrict__ mask, int ending) {
  const int wave = threadIdx.x >> 5, lane = threadIdx.x & 31;
  const long row = (long)blockIdx.x * 4 + wave;
  if (row >= (long)I_DIM * HT * I_DIM) return;
  const int m = (int)(row & 255), h = (int)((row >> 8) & 3), bi = (int)(row >> 10);
  u16* sr = S + row * 256;
  float v[8]; float mx = -3.0e38f;
#pragma unroll
  for (int t = 0; t < 8; ++t) {
    const int k = lane + t * 32;
    const float bias = ending ? Bh[((long)k * I_DIM + m) * 16 + h]
                              : Bh[((long)m * I_DIM + k) * 16 + h];
    const float mk = ending ? mask[k * I_DIM + bi] : mask[bi * I_DIM + k];
    const float x = b2f(sr[k]) + bias + (1.f - mk) * (-1e9f);
    v[t] = x; mx = fmaxf(mx, x);
  }
  for (int o = 16; o; o >>= 1) mx = fmaxf(mx, __shfl_xor(mx, o, 32));
  float s = 0.f;
#pragma unroll
  for (int t = 0; t < 8; ++t) { v[t] = __expf(v[t] - mx); s += v[t]; }
  for (int o = 16; o; o >>= 1) s += __shfl_xor(s, o, 32);
  const float inv = 1.f / s;
#pragma unroll
  for (int t = 0; t < 8; ++t) sr[lane + t * 32] = f2b(v[t] * inv);
}

// ------------- build Vt[(bi,h), d, k] from V[(i or k, j or k), h, d] --------
__global__ __launch_bounds__(256)
void tri_vt_pack_kernel(const u16* __restrict__ V, u16* __restrict__ Vt, int ending) {
  const int idx = blockIdx.x * 256 + threadIdx.x;
  if (idx >= I_DIM * HT * DT * I_DIM) return;
  const int k = idx & 255, d = (idx >> 8) & 31, bh = idx >> 13;
  const int bi = bh >> 2, h = bh & 3;
  const long src = ending ? (((long)k * I_DIM + bi) * CZ + h * DT + d)
                          : (((long)bi * I_DIM + k) * CZ + h * DT + d);
  Vt[idx] = V[src];
}

// ------------- gate+pack: gated[(i,j), h*32+d] = sigmoid(G)*vals ------------
__global__ __launch_bounds__(256)
void tri_gate_pack_kernel(const u16* __restrict__ G, const float* __restrict__ vals,
                          u16* __restrict__ out, int ending) {
  const int idx = blockIdx.x * 256 + threadIdx.x;
  if (idx >= I_DIM * HT * I_DIM * DT) return;
  const int d = idx & 31, m = (idx >> 5) & 255, bh = idx >> 13;
  const int bi = bh >> 2, h = bh & 3;
  const long row = ending ? ((long)m * I_DIM + bi) : ((long)bi * I_DIM + m);
  out[row * CZ + h * DT + d] = f2b(b2f(G[row * CZ + h * DT + d]) * vals[idx]);
}

// ============================================================================
// Host orchestration
// ============================================================================
static void gemm(hipStream_t st, const u16* A, const u16* B, const float* bias,
                 const float* rs, const float* resid, float* Cf, u16* Cb,
                 int M, int N, int K, long lda, long ldb, long ldc,
                 long sA0, long sA1, long sB0, long sB1, long sC0, long sC1,
                 int nb0, int nb1, float alpha, int act, int mode) {
  const int nt = (N % 256 == 0) ? 4 : ((N % 128 == 0) ? 2 : 1);
  dim3 grid((N + 64 * nt - 1) / (64 * nt), M / 16, nb0 * nb1);
  if (nt == 4)
    wmma_gemm_kernel<4><<<grid, 128, 0, st>>>(A, B, bias, rs, resid, Cf, Cb, M, N, K,
                                              lda, ldb, ldc, sA0, sA1, sB0, sB1,
                                              sC0, sC1, nb1, alpha, act, mode);
  else if (nt == 2)
    wmma_gemm_kernel<2><<<grid, 128, 0, st>>>(A, B, bias, rs, resid, Cf, Cb, M, N, K,
                                              lda, ldb, ldc, sA0, sA1, sB0, sB1,
                                              sC0, sC1, nb1, alpha, act, mode);
  else
    wmma_gemm_kernel<1><<<grid, 128, 0, st>>>(A, B, bias, rs, resid, Cf, Cb, M, N, K,
                                              lda, ldb, ldc, sA0, sA1, sB0, sB1,
                                              sC0, sC1, nb1, alpha, act, mode);
}
static void gemm1(hipStream_t st, const u16* A, const u16* B, const float* bias,
                  const float* rs, const float* resid, float* Cf, u16* Cb,
                  int M, int N, int K, float alpha, int act, int mode) {
  gemm(st, A, B, bias, rs, resid, Cf, Cb, M, N, K, K, K, N,
       0, 0, 0, 0, 0, 0, 1, 1, alpha, act, mode);
}

#define EW(kern, n, ...) kern<<<dim3(((n) + 255) / 256), dim3(256), 0, stream>>>(__VA_ARGS__)
#define ROWS(kern, r, ...) kern<<<dim3(((r) + 3) / 4), dim3(128), 0, stream>>>(__VA_ARGS__)

extern "C" void kernel_launch(void* const* d_in, const int* in_sizes, int n_in,
                              void* d_out, int out_size, void* d_ws, size_t ws_size,
                              hipStream_t stream) {
  const float* msa       = (const float*)d_in[0];
  const float* pair      = (const float*)d_in[1];
  const float* msa_mask  = (const float*)d_in[2];
  const float* pair_mask = (const float*)d_in[3];
  // params flattened (dict keys sorted): ga, mt, opm, pt, tri_end, tri_start
  const float* ga_bg = (const float*)d_in[4];
  const float* ga_bo = (const float*)d_in[5];
  const float* ga_lnb = (const float*)d_in[6];
  const float* ga_lng = (const float*)d_in[7];
  const float* ga_wg = (const float*)d_in[8];
  const float* ga_wk = (const float*)d_in[9];
  const float* ga_wo = (const float*)d_in[10];
  const float* ga_wq = (const float*)d_in[11];
  const float* ga_wv = (const float*)d_in[12];
  const float* mt_bd = (const float*)d_in[13];
  const float* mt_bu = (const float*)d_in[14];
  const float* mt_lnb = (const float*)d_in[15];
  const float* mt_lng = (const float*)d_in[16];
  const float* mt_wd = (const float*)d_in[17];
  const float* mt_wu = (const float*)d_in[18];
  const float* op_bl = (const float*)d_in[19];
  const float* op_bo = (const float*)d_in[20];
  const float* op_br = (const float*)d_in[21];
  const float* op_lnb = (const float*)d_in[22];
  const float* op_lng = (const float*)d_in[23];
  const float* op_wl = (const float*)d_in[24];
  const float* op_wo = (const float*)d_in[25];
  const float* op_wr = (const float*)d_in[26];
  const float* pt_bd = (const float*)d_in[27];
  const float* pt_bu = (const float*)d_in[28];
  const float* pt_lnb = (const float*)d_in[29];
  const float* pt_lng = (const float*)d_in[30];
  const float* pt_wd = (const float*)d_in[31];
  const float* pt_wu = (const float*)d_in[32];
  const float* te_bg = (const float*)d_in[33];
  const float* te_bo = (const float*)d_in[34];
  const float* te_lnb = (const float*)d_in[35];
  const float* te_lng = (const float*)d_in[36];
  const float* te_wb = (const float*)d_in[37];
  const float* te_wg = (const float*)d_in[38];
  const float* te_wk = (const float*)d_in[39];
  const float* te_wo = (const float*)d_in[40];
  const float* te_wq = (const float*)d_in[41];
  const float* te_wv = (const float*)d_in[42];
  const float* ts_bg = (const float*)d_in[43];
  const float* ts_bo = (const float*)d_in[44];
  const float* ts_lnb = (const float*)d_in[45];
  const float* ts_lng = (const float*)d_in[46];
  const float* ts_wb = (const float*)d_in[47];
  const float* ts_wg = (const float*)d_in[48];
  const float* ts_wk = (const float*)d_in[49];
  const float* ts_wo = (const float*)d_in[50];
  const float* ts_wq = (const float*)d_in[51];
  const float* ts_wv = (const float*)d_in[52];

  float* msa_out  = (float*)d_out;                     // [SI, CE] f32
  float* pair_out = msa_out + (size_t)SI * CE;         // [II, CZ] f32

  // ---- bump allocator over d_ws ----
  char* wbase = (char*)d_ws;
  size_t woff = 0;
  auto alloc = [&](size_t bytes) -> void* {
    void* p = wbase + woff;
    woff += (bytes + 255) & ~(size_t)255;
    return p;
  };
  float* pairA = (float*)alloc((size_t)II * CZ * 4);   // persistent pair ping
  float* pairB = (float*)alloc((size_t)II * CZ * 4);   // persistent pair pong
  float* msa1  = (float*)alloc((size_t)SI * CE * 4);   // msa after global attn
  const size_t scratch0 = woff;

  // ============ Phase A: MSA global attention ============
  {
    woff = scratch0;
    u16* xln = (u16*)alloc((size_t)SI * CE * 2);
    ROWS(ln_rows_kernel, SI, msa, ga_lng, ga_lnb, xln, SI, CE);
    u16* wk_p = (u16*)alloc(CE * CE * 2);
    u16* wv_p = (u16*)alloc(CE * CE * 2);
    u16* wg_p = (u16*)alloc(CE * CE * 2);
    u16* wq_p = (u16*)alloc(CE * CE * 2);
    u16* wo_p = (u16*)alloc(CE * CE * 2);
    EW(pack_w_kernel, CE * CE, ga_wk, CE, CE, CE, wk_p);
    EW(pack_w_kernel, CE * CE, ga_wv, CE, CE, CE, wv_p);
    EW(pack_w_kernel, CE * CE, ga_wg, CE, CE, CE, wg_p);
    EW(pack_w_kernel, CE * CE, ga_wq, CE, CE, CE, wq_p);
    EW(pack_w_kernel, CE * CE, ga_wo, CE, CE, CE, wo_p);

    u16* Kb = (u16*)alloc((size_t)SI * CE * 2);
    u16* Vb = (u16*)alloc((size_t)SI * CE * 2);
    u16* Gb = (u16*)alloc((size_t)SI * CE * 2);
    gemm1(stream, xln, wk_p, nullptr, nullptr, nullptr, nullptr, Kb, SI, CE, CE, 1.f, 0, 0);
    gemm1(stream, xln, wv_p, nullptr, nullptr, nullptr, nullptr, Vb, SI, CE, CE, 1.f, 0, 0);
    gemm1(stream, xln, wg_p, ga_bg, nullptr, nullptr, nullptr, Gb, SI, CE, CE, 1.f, 2, 0);

    float* xmean = (float*)alloc((size_t)I_DIM * CE * 4);
    EW(mean_s_kernel, I_DIM * CE, msa, xmean);
    u16* xmln = (u16*)alloc((size_t)I_DIM * CE * 2);
    ROWS(ln_rows_kernel, I_DIM, xmean, ga_lng, ga_lnb, xmln, I_DIM, CE);
    float* Qf = (float*)alloc((size_t)I_DIM * CE * 4);
    gemm1(stream, xmln, wq_p, nullptr, nullptr, nullptr, Qf, nullptr, I_DIM, CE, CE, 1.f, 0, 0);

    float* sc = (float*)alloc((size_t)I_DIM * HG * S_DIM * 4);
    EW(ga_score_kernel, S_DIM * I_DIM * HG, Qf, Kb, msa_mask, sc);
    ROWS(softmax512_kernel, I_DIM * HG, sc, I_DIM * HG);
    float* vals = (float*)alloc((size_t)I_DIM * CE * 4);
    EW(ga_wv_kernel, I_DIM * HG * DG, sc, Vb, vals);
    u16* gated = (u16*)alloc((size_t)SI * CE * 2);
    EW(ga_gate_kernel, SI * CE, Gb, vals, gated);
    // out = msa + mask * (gated @ wo + bo)
    gemm1(stream, gated, wo_p, ga_bo, msa_mask, msa, msa1, nullptr, SI, CE, CE, 1.f, 0, 1);
  }

  // ============ Phase B: MSA transition -> msa_out ============
  {
    woff = scratch0;
    u16* xln = (u16*)alloc((size_t)SI * CE * 2);
    ROWS(ln_rows_kernel, SI, msa1, mt_lng, mt_lnb, xln, SI, CE);
    u16* wu_p = (u16*)alloc(256 * CE * 2);
    u16* wd_p = (u16*)alloc(CE * 256 * 2);
    EW(pack_w_kernel, 256 * CE, mt_wu, CE, 256, 256, wu_p);
    EW(pack_w_kernel, CE * 256, mt_wd, 256, CE, CE, wd_p);
    u16* hb = (u16*)alloc((size_t)SI * 256 * 2);
    gemm1(stream, xln, wu_p, mt_bu, nullptr, nullptr, nullptr, hb, SI, 256, CE, 1.f, 1, 0);
    gemm1(stream, hb, wd_p, mt_bd, nullptr, msa1, msa_out, nullptr, SI, CE, 256, 1.f, 0, 0);
  }

  // ============ Phase C: outer product mean -> pairA ============
  {
    woff = scratch0;
    u16* xln = (u16*)alloc((size_t)SI * CE * 2);
    ROWS(ln_rows_kernel, SI, msa_out, op_lng, op_lnb, xln, SI, CE);
    u16* wl_p = (u16*)alloc(COP * CE * 2);
    u16* wr_p = (u16*)alloc(COP * CE * 2);
    EW(pack_w_kernel, COP * CE, op_wl, CE, COP, COP, wl_p);
    EW(pack_w_kernel, COP * CE, op_wr, CE, COP, COP, wr_p);
    u16* Ab = (u16*)alloc((size_t)SI * COP * 2);
    u16* Bb = (u16*)alloc((size_t)SI * COP * 2);
    gemm1(stream, xln, wl_p, op_bl, msa_mask, nullptr, nullptr, Ab, SI, COP, CE, 1.f, 0, 1);
    gemm1(stream, xln, wr_p, op_br, msa_mask, nullptr, nullptr, Bb, SI, COP, CE, 1.f, 0, 1);
    u16* At = (u16*)alloc((size_t)8192 * S_DIM * 2);
    u16* Bt = (u16*)alloc((size_t)8192 * S_DIM * 2);
    EW(opm_tpack_kernel, 8192 * S_DIM, Ab, At);
    EW(opm_tpack_kernel, 8192 * S_DIM, Bb, Bt);
    // outer[(i,c),(j,d)] = At @ Bt^T   (8192 x 8192 x 512)
    u16* O = (u16*)alloc((size_t)8192 * 8192 * 2);
    gemm1(stream, At, Bt, nullptr, nullptr, nullptr, nullptr, O, 8192, 8192, S_DIM, 1.f, 0, 0);
    u16* O2 = (u16*)alloc((size_t)II * 1024 * 2);
    EW(opm_repack_kernel, II * 1024, O, O2);
    float* ninv = (float*)alloc((size_t)II * 4);
    EW(norm_inv_kernel, II, msa_mask, ninv);
    u16* wo_p = (u16*)alloc(CZ * 1024 * 2);
    EW(pack_w_kernel, CZ * 1024, op_wo, 1024, CZ, CZ, wo_p);
    // pairA = pair + (O2 @ wo) * ninv[m] + bo
    gemm1(stream, O2, wo_p, op_bo, ninv, pair, pairA, nullptr, II, CZ, 1024, 1.f, 0, 2);
  }

  // ============ Phases D/E: triangle attention (start, end) ============
  auto run_tri = [&](const float* lng, const float* lnb, const float* wq,
                     const float* wk, const float* wv, const float* wb,
                     const float* wg, const float* bg, const float* wo,
                     const float* bo, int ending, const float* zin, float* zout) {
    woff = scratch0;
    u16* zl = (u16*)alloc((size_t)II * CZ * 2);
    ROWS(ln_rows_kernel, II, zin, lng, lnb, zl, II, CZ);
    u16* wq_p = (u16*)alloc(CZ * CZ * 2);
    u16* wk_p = (u16*)alloc(CZ * CZ * 2);
    u16* wv_p = (u16*)alloc(CZ * CZ * 2);
    u16* wg_p = (u16*)alloc(CZ * CZ * 2);
    u16* wo_p = (u16*)alloc(CZ * CZ * 2);
    u16* wb_p = (u16*)alloc(16 * CZ * 2);
    EW(pack_w_kernel, CZ * CZ, wq, CZ, CZ, CZ, wq_p);
    EW(pack_w_kernel, CZ * CZ, wk, CZ, CZ, CZ, wk_p);
    EW(pack_w_kernel, CZ * CZ, wv, CZ, CZ, CZ, wv_p);
    EW(pack_w_kernel, CZ * CZ, wg, CZ, CZ, CZ, wg_p);
    EW(pack_w_kernel, CZ * CZ, wo, CZ, CZ, CZ, wo_p);
    EW(pack_w_kernel, 16 * CZ, wb, CZ, HT, 16, wb_p);    // zero-padded N: 4 -> 16

    u16* Qb = (u16*)alloc((size_t)II * CZ * 2);
    u16* Kb = (u16*)alloc((size_t)II * CZ * 2);
    u16* Vb = (u16*)alloc((size_t)II * CZ * 2);
    u16* Gb = (u16*)alloc((size_t)II * CZ * 2);
    float* Bhf = (float*)alloc((size_t)II * 16 * 4);
    gemm1(stream, zl, wq_p, nullptr, nullptr, nullptr, nullptr, Qb, II, CZ, CZ, 1.f, 0, 0);
    gemm1(stream, zl, wk_p, nullptr, nullptr, nullptr, nullptr, Kb, II, CZ, CZ, 1.f, 0, 0);
    gemm1(stream, zl, wv_p, nullptr, nullptr, nullptr, nullptr, Vb, II, CZ, CZ, 1.f, 0, 0);
    gemm1(stream, zl, wg_p, bg, nullptr, nullptr, nullptr, Gb, II, CZ, CZ, 1.f, 2, 0);
    gemm1(stream, zl, wb_p, nullptr, nullptr, nullptr, Bhf, nullptr, II, 16, CZ, 1.f, 0, 0);

    // scores: batch (bi, h), 256x256x32, alpha = 1/sqrt(32)
    u16* Sb = (u16*)alloc((size_t)I_DIM * HT * II * 2);
    const long sQbi = ending ? CZ : (long)I_DIM * CZ;   // start: i-stride; end: j-stride
    const long ldQ  = ending ? (long)I_DIM * CZ : CZ;
    gemm(stream, Qb, Kb, nullptr, nullptr, nullptr, nullptr, Sb,
         I_DIM, I_DIM, DT, ldQ, ldQ, I_DIM,
         sQbi, DT, sQbi, DT, (long)HT * II, II,
         I_DIM, HT, 0.17677670f, 0, 0);
    tri_softmax_kernel<<<dim3((I_DIM * HT * I_DIM + 3) / 4), dim3(128), 0, stream>>>(
        Sb, Bhf, pair_mask, ending);

    u16* Vt = (u16*)alloc((size_t)I_DIM * HT * DT * I_DIM * 2);
    EW(tri_vt_pack_kernel, I_DIM * HT * DT * I_DIM, Vb, Vt, ending);
    float* valsf = (float*)alloc((size_t)I_DIM * HT * I_DIM * DT * 4);
    // vals = attn @ Vt^T : batch (bi, h), 256x32x256
    gemm(stream, Sb, Vt, nullptr, nullptr, nullptr, valsf, nullptr,
         I_DIM, DT, I_DIM, I_DIM, I_DIM, DT,
         (long)HT * II, II, (long)HT * DT * I_DIM, (long)DT * I_DIM,
         (long)HT * I_DIM * DT, (long)I_DIM * DT,
         I_DIM, HT, 1.f, 0, 0);

    u16* gated = (u16*)alloc((size_t)II * CZ * 2);
    EW(tri_gate_pack_kernel, I_DIM * HT * I_DIM * DT, Gb, valsf, gated, ending);
    // zout = zin + mask * (gated @ wo + bo)
    gemm1(stream, gated, wo_p, bo, pair_mask, zin, zout, nullptr, II, CZ, CZ, 1.f, 0, 1);
  };

  run_tri(ts_lng, ts_lnb, ts_wq, ts_wk, ts_wv, ts_wb, ts_wg, ts_bg, ts_wo, ts_bo,
          0, pairA, pairB);
  run_tri(te_lng, te_lnb, te_wq, te_wk, te_wv, te_wb, te_wg, te_bg, te_wo, te_bo,
          1, pairB, pairA);

  // ============ Phase F: pair transition -> pair_out ============
  {
    woff = scratch0;
    u16* zl = (u16*)alloc((size_t)II * CZ * 2);
    ROWS(ln_rows_kernel, II, pairA, pt_lng, pt_lnb, zl, II, CZ);
    u16* wu_p = (u16*)alloc(512 * CZ * 2);
    u16* wd_p = (u16*)alloc(CZ * 512 * 2);
    EW(pack_w_kernel, 512 * CZ, pt_wu, CZ, 512, 512, wu_p);
    EW(pack_w_kernel, CZ * 512, pt_wd, 512, CZ, CZ, wd_p);
    u16* hb = (u16*)alloc((size_t)II * 512 * 2);
    gemm1(stream, zl, wu_p, pt_bu, nullptr, nullptr, nullptr, hb, II, 512, CZ, 1.f, 1, 0);
    gemm1(stream, hb, wd_p, pt_bd, nullptr, pairA, pair_out, nullptr, II, CZ, 512, 1.f, 0, 0);
  }

  (void)in_sizes; (void)n_in; (void)out_size; (void)ws_size;
}
